// Autoencoder_VQVAE_1451698946500
// MI455X (gfx1250) — matrix operand order repaired and checked by reading, not compiled, not run on hardware
//
#include <hip/hip_runtime.h>
#include <hip/hip_bf16.h>
#include <stdint.h>

// ---- problem constants ----
#define TT 30
#define BB 1024
#define DP 135
#define HH 512
#define KC 512
#define OUT_ELEMS (1024*30*135)   // 4,147,200

typedef __attribute__((ext_vector_type(16))) __bf16 v16bf;
typedef __attribute__((ext_vector_type(8)))  float  v8f;

__device__ __forceinline__ unsigned short f2bf(float f) {
  union { float f; unsigned u; } x; x.f = f;
  unsigned r = x.u + 0x7FFFu + ((x.u >> 16) & 1u);   // RNE
  return (unsigned short)(r >> 16);
}
// native __bf16 conversion: lets clang select hardware v_cvt_*bf16* if present
__device__ __forceinline__ unsigned pack2(float a, float b) {
  union { __bf16 h[2]; unsigned u; } t;
  t.h[0] = (__bf16)a;
  t.h[1] = (__bf16)b;
  return t.u;
}

// =====================================================================
// WMMA GEMM:  C[M,Nreal] = remap(A) * Bm (+ bias)
//  Bm: bf16, k-tiled layout  Bm[((k>>5)*Npad + n)*32 + (k&31)]
//  AMODE 0: src = A[m*lda + kk]                     (kk < KrealA else 0)
//  AMODE 1: encoder conv im2col: m=(l,b); kk=(chunk,i)
//           src = A[((l*Kc+chunk)*BB + b)*HH + i]
//  AMODE 2: decoder convT: m=(l',b); l=l'/Kc; k=l'%Kc selects B slice
//           src = A[(l*BB + b)*HH + kk]
//  Block 256 thr = 8 waves; tile 64(M) x 128(N); K step 32; double-buffered.
//  Requires: M % 64 == 0, Npad % 128 == 0 over launched grid, rows-per-l = BB.
// =====================================================================
template <int AMODE>
__global__ __launch_bounds__(256)
void gemm_wmma(const float* __restrict__ A, const unsigned short* __restrict__ Bm,
               float* __restrict__ C, const float* __restrict__ bias,
               int M, int Nreal, int Npad, int KK, int KrealA,
               int lda, int Kc, int ldc)
{
  __shared__ __align__(16) unsigned short lA[2][64 * 32];
  __shared__ __align__(16) unsigned short lB[2][128 * 32];

  const int tid  = threadIdx.x;
  const int lane = tid & 31;
  const int wave = tid >> 5;
  const int wr   = wave >> 1;           // M sub-tile 0..3
  const int wc   = wave & 1;            // N sub-tile 0..1
  const int m0   = blockIdx.y * 64;
  const int n0   = blockIdx.x * 128;

  const unsigned short* Bp = Bm;
  if (AMODE == 2) {
    int lp = m0 >> 10;                   // /BB, uniform across 64-row tile
    Bp += (size_t)(lp % Kc) * (size_t)KK * (size_t)Npad;
  }

  // A staging assignment: thread -> (row r, k-octet jb)
  const int r  = tid >> 2;               // 0..63
  const int jb = (tid & 3) * 8;          // 0,8,16,24
  const int m  = m0 + r;
  const int nk = KK >> 5;

  // loop-invariant A row base
  size_t base0;  size_t chStride = 0;
  bool   vecA = true;
  if (AMODE == 0) {
    base0 = (size_t)m * lda;
    vecA  = (((lda & 3) == 0) && (KrealA == KK));
  } else if (AMODE == 1) {
    int l = m >> 10, b = m & (BB - 1);
    base0    = ((size_t)l * Kc * BB + b) * HH;
    chStride = (size_t)BB * HH;
  } else {
    int lp = m >> 10, b = m & (BB - 1);
    base0 = ((size_t)(lp / Kc) * BB + b) * HH;
  }

  uint4 apack, bq0, bq1;

  auto loadStage = [&](int kt) {
    const int kkk = (kt << 5) + jb;
    // ---- A: 8 contiguous fp32 -> 8 bf16 ----
    if (AMODE != 0 || vecA) {
      size_t base;
      if (AMODE == 0)      base = base0 + kkk;
      else if (AMODE == 1) base = base0 + (size_t)(kkk >> 9) * chStride + (kkk & 511);
      else                 base = base0 + kkk;
      const float4 v0 = *(const float4*)(A + base);
      const float4 v1 = *(const float4*)(A + base + 4);
      apack.x = pack2(v0.x, v0.y); apack.y = pack2(v0.z, v0.w);
      apack.z = pack2(v1.x, v1.y); apack.w = pack2(v1.z, v1.w);
    } else {
      float f[8];
      #pragma unroll
      for (int j = 0; j < 8; ++j) {
        int kx = kkk + j;
        f[j] = (kx < KrealA) ? A[base0 + kx] : 0.f;
      }
      apack.x = pack2(f[0], f[1]); apack.y = pack2(f[2], f[3]);
      apack.z = pack2(f[4], f[5]); apack.w = pack2(f[6], f[7]);
    }
    // ---- B: contiguous 128x32 bf16 tile (8 KB) ----
    const uint4* s = (const uint4*)(Bp + ((size_t)kt * Npad + n0) * 32);
    bq0 = s[tid];
    bq1 = s[tid + 256];
  };
  auto storeStage = [&](int pb) {
    *(uint4*)&lA[pb][r * 32 + jb] = apack;
    uint4* d = (uint4*)lB[pb];
    d[tid]       = bq0;
    d[tid + 256] = bq1;
  };

  v8f acc[4];
  #pragma unroll
  for (int s = 0; s < 4; ++s) acc[s] = (v8f){0.f,0.f,0.f,0.f,0.f,0.f,0.f,0.f};

  const int arow  = wr * 16 + (lane & 15);
  const int kaddA = (lane & 16) ? 8 : 0;      // A: two octets at kaddA, kaddA+16
  const int bcol  = lane & 15;
  const int kaddB = (lane & 16) ? 8 : 0;      // B: two octets at kaddB, kaddB+16

  auto compute = [&](int pb) {
    union { v16bf v; uint4 q[2]; } af;
    af.q[0] = *(const uint4*)&lA[pb][arow * 32 + kaddA];
    af.q[1] = *(const uint4*)&lA[pb][arow * 32 + kaddA + 16];
    #pragma unroll
    for (int s = 0; s < 4; ++s) {
      const int col = wc * 64 + s * 16 + bcol;
      union { v16bf v; uint4 q[2]; } bfm;
      bfm.q[0] = *(const uint4*)&lB[pb][col * 32 + kaddB];
      bfm.q[1] = *(const uint4*)&lB[pb][col * 32 + kaddB + 16];
      acc[s] = __builtin_amdgcn_wmma_f32_16x16x32_bf16(
          false, af.v, false, bfm.v, (short)0, acc[s], false, false);
    }
  };

  loadStage(0);
  storeStage(0);
  // steady state: branch-free double-buffered pipeline
  for (int kt = 0; kt < nk - 1; ++kt) {
    __syncthreads();
    loadStage(kt + 1);
    compute(kt & 1);
    storeStage((kt + 1) & 1);
  }
  __syncthreads();
  compute((nk - 1) & 1);

  // ---- store (C/D layout: VGPR v -> M = v + (lane>=16 ? 8:0)) ----
  const int rbase = m0 + wr * 16 + ((lane & 16) ? 8 : 0);
  const int cbase = n0 + wc * 64 + (lane & 15);
  #pragma unroll
  for (int s = 0; s < 4; ++s) {
    int col = cbase + s * 16;
    if (col < Nreal) {
      float bv = bias ? bias[col] : 0.f;
      #pragma unroll
      for (int v = 0; v < 8; ++v) {
        int row = rbase + v;
        if (row < M) C[(size_t)row * ldc + col] = acc[s][v] + bv;
      }
    }
  }
}

// =====================================================================
// Weight repacks: fp32 -> bf16, k-tiled layout ((k>>5)*Npad + n)*32 + (k&31)
// =====================================================================
__device__ __forceinline__ size_t btile(int k, int n, int Npad) {
  return ((size_t)(k >> 5) * Npad + n) * 32 + (k & 31);
}
__global__ void repack_plain(const float* __restrict__ W, unsigned short* __restrict__ Bm,
                             int Kreal, int Kpad, int Nreal, int Npad) {
  size_t tot = (size_t)Kpad * Npad;
  for (size_t i = blockIdx.x * blockDim.x + threadIdx.x; i < tot; i += (size_t)gridDim.x * blockDim.x) {
    int n = (int)(i % Npad); int k = (int)(i / Npad);
    float v = (k < Kreal && n < Nreal) ? W[(size_t)k * Nreal + n] : 0.f;
    Bm[btile(k, n, Npad)] = f2bf(v);
  }
}
// encoder conv w (O,I,Kc): GEMM row kk = kc*I+i, col o
__global__ void repack_encw(const float* __restrict__ W, unsigned short* __restrict__ Bm,
                            int O, int I, int Kc) {
  size_t tot = (size_t)Kc * I * O;
  for (size_t idx = blockIdx.x * blockDim.x + threadIdx.x; idx < tot; idx += (size_t)gridDim.x * blockDim.x) {
    int o = (int)(idx % O); size_t r = idx / O;
    int i = (int)(r % I);   int kc = (int)(r / I);
    Bm[btile(kc * I + i, o, O)] = f2bf(W[((size_t)o * I + i) * Kc + kc]);
  }
}
// decoder convT w (I,O,Kc): slice kc, GEMM row i, col o
__global__ void repack_decw(const float* __restrict__ W, unsigned short* __restrict__ Bm,
                            int O, int I, int Kc) {
  size_t tot = (size_t)Kc * I * O;
  for (size_t idx = blockIdx.x * blockDim.x + threadIdx.x; idx < tot; idx += (size_t)gridDim.x * blockDim.x) {
    int o = (int)(idx % O); size_t r = idx / O;
    int i = (int)(r % I);   int kc = (int)(r / I);
    Bm[(size_t)kc * I * O + btile(i, o, O)] = f2bf(W[((size_t)i * O + o) * Kc + kc]);
  }
}
// codebook (KC,HH) -> transposed bf16 (HH x KC), k-tiled
__global__ void repack_cbT(const float* __restrict__ Cb, unsigned short* __restrict__ Bm) {
  size_t tot = (size_t)HH * KC;
  for (size_t idx = blockIdx.x * blockDim.x + threadIdx.x; idx < tot; idx += (size_t)gridDim.x * blockDim.x) {
    int code = (int)(idx % KC); int i = (int)(idx / KC);
    Bm[btile(i, code, KC)] = f2bf(Cb[(size_t)code * HH + i]);
  }
}

// =====================================================================
// BatchNorm helpers over (L,B,H) layout: stats per column (channel)
// =====================================================================
__global__ __launch_bounds__(256)
void col_stats(const float* __restrict__ X, int M, int rowsPerBlock, float* __restrict__ sums) {
  int tid = threadIdx.x;
  int c0 = tid, c1 = tid + 256;
  int r0 = blockIdx.x * rowsPerBlock;
  int r1 = min(M, r0 + rowsPerBlock);
  float s0 = 0.f, s1 = 0.f, q0 = 0.f, q1 = 0.f;
  for (int r = r0; r < r1; ++r) {
    float a = X[(size_t)r * HH + c0];
    float b = X[(size_t)r * HH + c1];
    s0 += a; q0 += a * a;
    s1 += b; q1 += b * b;
  }
  atomicAdd(&sums[c0], s0);       atomicAdd(&sums[c1], s1);
  atomicAdd(&sums[HH + c0], q0);  atomicAdd(&sums[HH + c1], q1);
}
__global__ void bn_coef(const float* __restrict__ sums, const float* __restrict__ g,
                        const float* __restrict__ b, float* __restrict__ coef, float invM) {
  int c = blockIdx.x * blockDim.x + threadIdx.x;
  if (c < HH) {
    float m = sums[c] * invM;
    float v = sums[HH + c] * invM - m * m;
    float a = g[c] * rsqrtf(v + 1e-5f);
    coef[c] = a;
    coef[HH + c] = b[c] - a * m;
  }
}
__global__ void bn_apply_relu(float* __restrict__ X, size_t n, const float* __restrict__ coef) {
  for (size_t i = blockIdx.x * blockDim.x + threadIdx.x; i < n; i += (size_t)gridDim.x * blockDim.x) {
    int c = (int)(i & (HH - 1));
    X[i] = fmaxf(coef[c] * X[i] + coef[HH + c], 0.f);
  }
}

// =====================================================================
// VQ kernels
// =====================================================================
__global__ __launch_bounds__(128)
void code_norms(const float* __restrict__ Cb, float* __restrict__ cn) {
  __shared__ float sh[128];
  int code = blockIdx.x, tid = threadIdx.x;
  float s = 0.f;
  for (int i = tid; i < HH; i += 128) { float v = Cb[(size_t)code * HH + i]; s += v * v; }
  sh[tid] = s; __syncthreads();
  for (int w = 64; w > 0; w >>= 1) { if (tid < w) sh[tid] += sh[tid + w]; __syncthreads(); }
  if (tid == 0) cn[code] = sh[0];
}
__global__ __launch_bounds__(256)
void vq_argmin(const float* __restrict__ dots, const float* __restrict__ cn, int* __restrict__ idx) {
  __shared__ float sv[256]; __shared__ int si[256];
  int row = blockIdx.x, tid = threadIdx.x;
  float best = 3.4e38f; int bi = 0x7fffffff;
  for (int j = tid; j < KC; j += 256) {
    float v = cn[j] - 2.f * dots[(size_t)row * KC + j];
    if (v < best || (v == best && j < bi)) { best = v; bi = j; }
  }
  sv[tid] = best; si[tid] = bi; __syncthreads();
  for (int w = 128; w > 0; w >>= 1) {
    if (tid < w) {
      float ov = sv[tid + w]; int oi = si[tid + w];
      if (ov < sv[tid] || (ov == sv[tid] && oi < si[tid])) { sv[tid] = ov; si[tid] = oi; }
    }
    __syncthreads();
  }
  if (tid == 0) idx[row] = si[0];
}
__global__ __launch_bounds__(256)
void vq_gather_loss(const float* __restrict__ flat, const float* __restrict__ Cb,
                    const int* __restrict__ idx, float* __restrict__ quant,
                    float* __restrict__ lossAcc, float* __restrict__ counts) {
  __shared__ float sh[256];
  int row = blockIdx.x, tid = threadIdx.x;
  int code = idx[row];
  float acc = 0.f;
  for (int c = tid; c < HH; c += 256) {
    float q = Cb[(size_t)code * HH + c];
    float x = flat[(size_t)row * HH + c];
    quant[(size_t)row * HH + c] = q;
    float d = q - x; acc += d * d;
  }
  sh[tid] = acc; __syncthreads();
  for (int w = 128; w > 0; w >>= 1) { if (tid < w) sh[tid] += sh[tid + w]; __syncthreads(); }
  if (tid == 0) { atomicAdd(lossAcc, sh[0]); atomicAdd(&counts[code], 1.f); }
}
__global__ __launch_bounds__(512)
void vq_finalize(const float* __restrict__ lossAcc, const float* __restrict__ counts,
                 float* __restrict__ outp) {
  __shared__ float sh[512];
  int c = threadIdx.x;
  float p = counts[c] * (1.f / 1024.f);
  sh[c] = -p * logf(p + 1e-10f);
  __syncthreads();
  for (int w = 256; w > 0; w >>= 1) { if (c < w) sh[c] += sh[c + w]; __syncthreads(); }
  if (c == 0) {
    outp[OUT_ELEMS]     = 1.25f * lossAcc[0] * (1.f / (1024.f * 512.f));
    outp[OUT_ELEMS + 1] = expf(sh[0]);
  }
}

// final:  G[(l*BB+b)*135+p] + b_out[p]  ->  out[(b*30+l)*135+p]
__global__ void final_out(const float* __restrict__ G, const float* __restrict__ bout,
                          float* __restrict__ out) {
  size_t tot = (size_t)TT * BB * DP;
  for (size_t i = blockIdx.x * blockDim.x + threadIdx.x; i < tot; i += (size_t)gridDim.x * blockDim.x) {
    int p = (int)(i % DP); size_t m = i / DP;
    int l = (int)(m >> 10); int b = (int)(m & 1023);
    out[((size_t)b * TT + l) * DP + p] = G[i] + bout[p];
  }
}

// =====================================================================
// host launcher
// =====================================================================
static void launch_gemm(const float* A, const unsigned short* Bm, float* C, const float* bias,
                        int M, int Nreal, int Npad, int KK, int KrealA,
                        int amode, int lda, int Kc, int ldc, hipStream_t s) {
  dim3 grid((Nreal + 127) / 128, (M + 63) / 64);
  if (amode == 0)
    gemm_wmma<0><<<grid, 256, 0, s>>>(A, Bm, C, bias, M, Nreal, Npad, KK, KrealA, lda, Kc, ldc);
  else if (amode == 1)
    gemm_wmma<1><<<grid, 256, 0, s>>>(A, Bm, C, bias, M, Nreal, Npad, KK, KrealA, lda, Kc, ldc);
  else
    gemm_wmma<2><<<grid, 256, 0, s>>>(A, Bm, C, bias, M, Nreal, Npad, KK, KrealA, lda, Kc, ldc);
}
static void run_bn(float* X, int M, const float* g, const float* b,
                   float* stats, float* coef, hipStream_t s) {
  hipMemsetAsync(stats, 0, 2 * HH * sizeof(float), s);
  int rpb = 128;
  col_stats<<<(M + rpb - 1) / rpb, 256, 0, s>>>(X, M, rpb, stats);
  bn_coef<<<2, 256, 0, s>>>(stats, g, b, coef, 1.f / (float)M);
  size_t n = (size_t)M * HH;
  bn_apply_relu<<<1024, 256, 0, s>>>(X, n, coef);
}

extern "C" void kernel_launch(void* const* d_in, const int* in_sizes, int n_in,
                              void* d_out, int out_size, void* d_ws, size_t ws_size,
                              hipStream_t stream) {
  const float* input = (const float*)d_in[0];
  const float* W_in  = (const float*)d_in[1];  const float* b_in  = (const float*)d_in[2];
  const float* w_e1  = (const float*)d_in[3];  const float* g_e1  = (const float*)d_in[4];  const float* be1 = (const float*)d_in[5];
  const float* w_e2  = (const float*)d_in[6];  const float* g_e2  = (const float*)d_in[7];  const float* be2 = (const float*)d_in[8];
  const float* w_e3  = (const float*)d_in[9];  const float* g_e3  = (const float*)d_in[10]; const float* be3 = (const float*)d_in[11];
  const float* cbk   = (const float*)d_in[12];
  const float* W_q   = (const float*)d_in[13]; const float* b_q   = (const float*)d_in[14];
  const float* w_d1  = (const float*)d_in[15]; const float* g_d1  = (const float*)d_in[16]; const float* bd1 = (const float*)d_in[17];
  const float* w_d2  = (const float*)d_in[18]; const float* g_d2  = (const float*)d_in[19]; const float* bd2 = (const float*)d_in[20];
  const float* w_d3  = (const float*)d_in[21]; const float* g_d3  = (const float*)d_in[22]; const float* bd3 = (const float*)d_in[23];
  const float* W_out = (const float*)d_in[24]; const float* b_out = (const float*)d_in[25];
  float* outp = (float*)d_out;

  char* ws = (char*)d_ws;
  size_t off = 0;
  auto alloc = [&](size_t bytes) -> void* {
    void* p = ws + off; off += (bytes + 255) & ~(size_t)255; return p;
  };
  const size_t MB = (size_t)TT * BB;                       // 30720
  float* bufA = (float*)alloc(MB * HH * sizeof(float));
  float* bufB = (float*)alloc(MB * HH * sizeof(float));
  float* gout = (float*)alloc(MB * DP * sizeof(float));
  unsigned short* wIn = (unsigned short*)alloc((size_t)160 * HH * 2);
  unsigned short* wE1 = (unsigned short*)alloc((size_t)5 * HH * HH * 2);
  unsigned short* wE2 = (unsigned short*)alloc((size_t)3 * HH * HH * 2);
  unsigned short* wE3 = (unsigned short*)alloc((size_t)2 * HH * HH * 2);
  unsigned short* wQ  = (unsigned short*)alloc((size_t)HH * HH * 2);
  unsigned short* wD1 = (unsigned short*)alloc((size_t)2 * HH * HH * 2);
  unsigned short* wD2 = (unsigned short*)alloc((size_t)3 * HH * HH * 2);
  unsigned short* wD3 = (unsigned short*)alloc((size_t)5 * HH * HH * 2);
  unsigned short* wO  = (unsigned short*)alloc((size_t)HH * 256 * 2);   // Npad=256
  unsigned short* cbT = (unsigned short*)alloc((size_t)HH * KC * 2);
  float* dots  = (float*)alloc((size_t)BB * KC * sizeof(float));
  float* quant = (float*)alloc((size_t)BB * HH * sizeof(float));
  int*   idx   = (int*)alloc((size_t)BB * sizeof(int));
  float* cn    = (float*)alloc((size_t)KC * sizeof(float));
  float* stats = (float*)alloc((size_t)2 * HH * sizeof(float));
  float* coef  = (float*)alloc((size_t)2 * HH * sizeof(float));
  float* lossA = (float*)alloc(sizeof(float));
  float* cnts  = (float*)alloc((size_t)KC * sizeof(float));
  (void)ws_size; (void)in_sizes; (void)n_in; (void)out_size;

  // ---- weight repacks (fp32 -> padded bf16, k-tiled) ----
  repack_plain<<<256, 256, 0, stream>>>(W_in, wIn, DP, 160, HH, HH);
  repack_encw <<<512, 256, 0, stream>>>(w_e1, wE1, HH, HH, 5);
  repack_encw <<<512, 256, 0, stream>>>(w_e2, wE2, HH, HH, 3);
  repack_encw <<<512, 256, 0, stream>>>(w_e3, wE3, HH, HH, 2);
  repack_plain<<<512, 256, 0, stream>>>(W_q, wQ, HH, HH, HH, HH);
  repack_decw <<<512, 256, 0, stream>>>(w_d1, wD1, HH, HH, 2);
  repack_decw <<<512, 256, 0, stream>>>(w_d2, wD2, HH, HH, 3);
  repack_decw <<<512, 256, 0, stream>>>(w_d3, wD3, HH, HH, 5);
  repack_plain<<<256, 256, 0, stream>>>(W_out, wO, HH, HH, DP, 256);
  repack_cbT  <<<512, 256, 0, stream>>>(cbk, cbT);

  // ---- encoder ----
  launch_gemm(input, wIn, bufA, b_in, (int)MB, HH, HH, 160, DP, 0, DP, 1, HH, stream);
  launch_gemm(bufA, wE1, bufB, nullptr, 6 * BB, HH, HH, 5 * HH, 5 * HH, 1, 0, 5, HH, stream);
  run_bn(bufB, 6 * BB, g_e1, be1, stats, coef, stream);
  launch_gemm(bufB, wE2, bufA, nullptr, 2 * BB, HH, HH, 3 * HH, 3 * HH, 1, 0, 3, HH, stream);
  run_bn(bufA, 2 * BB, g_e2, be2, stats, coef, stream);
  launch_gemm(bufA, wE3, bufB, nullptr, BB, HH, HH, 2 * HH, 2 * HH, 1, 0, 2, HH, stream);
  run_bn(bufB, BB, g_e3, be3, stats, coef, stream);

  // ---- VQ ----
  code_norms<<<KC, 128, 0, stream>>>(cbk, cn);
  launch_gemm(bufB, cbT, dots, nullptr, BB, KC, KC, HH, HH, 0, HH, 1, KC, stream);
  vq_argmin<<<BB, 256, 0, stream>>>(dots, cn, idx);
  hipMemsetAsync(lossA, 0, sizeof(float), stream);
  hipMemsetAsync(cnts, 0, KC * sizeof(float), stream);
  vq_gather_loss<<<BB, 256, 0, stream>>>(bufB, cbk, idx, quant, lossA, cnts);

  // ---- decoder ----
  launch_gemm(quant, wQ, bufA, b_q, BB, HH, HH, HH, HH, 0, HH, 1, HH, stream);
  launch_gemm(bufA, wD1, bufB, nullptr, 2 * BB, HH, HH, HH, HH, 2, 0, 2, HH, stream);
  run_bn(bufB, 2 * BB, g_d1, bd1, stats, coef, stream);
  launch_gemm(bufB, wD2, bufA, nullptr, 6 * BB, HH, HH, HH, HH, 2, 0, 3, HH, stream);
  run_bn(bufA, 6 * BB, g_d2, bd2, stats, coef, stream);
  launch_gemm(bufA, wD3, bufB, nullptr, (int)MB, HH, HH, HH, HH, 2, 0, 5, HH, stream);
  run_bn(bufB, (int)MB, g_d3, bd3, stats, coef, stream);

  // ---- output projection + layout fixup + scalars ----
  launch_gemm(bufB, wO, gout, nullptr, (int)MB, DP, 256, HH, HH, 0, HH, 1, DP, stream);
  final_out<<<2048, 256, 0, stream>>>(gout, b_out, outp);
  vq_finalize<<<1, 512, 0, stream>>>(lossA, cnts, outp);
}
